// TransformerXLModel_66915590472437
// MI455X (gfx1250) — matrix operand order repaired and checked by reading, not compiled, not run on hardware
//
#include <hip/hip_runtime.h>
#include <math.h>

// ---------------- problem constants ----------------
#define Q_LEN  512
#define BSZ    8
#define VOCAB  32000
#define DMODEL 1024
#define NHEAD  16
#define DHEAD  64
#define FFDIM  4096
#define NLAYER 4
#define NTOK   (Q_LEN * BSZ)   // 4096 tokens
#define RLEN   (Q_LEN + 1)     // 513 relative positions
#define BHB    (BSZ * NHEAD)   // 128 attention batches

typedef __attribute__((ext_vector_type(16))) __bf16 v16bf;
typedef __attribute__((ext_vector_type(8)))  __bf16 v8bf;
typedef __attribute__((ext_vector_type(4)))  __bf16 v4bf;
typedef __attribute__((ext_vector_type(2)))  __bf16 v2bf;
typedef __attribute__((ext_vector_type(8)))  float  v8f;
typedef __attribute__((ext_vector_type(4)))  float  f4;

// explicit global address space => global_load_* / global_store_* (not FLAT)
typedef const float __attribute__((address_space(1)))* gcf;
typedef       float __attribute__((address_space(1)))* gf;
typedef const f4    __attribute__((address_space(1)))* gcf4;

static __device__ __forceinline__ __bf16 f2bf(float f) {
  unsigned u = __builtin_bit_cast(unsigned, f);
  u += 0x7FFFu + ((u >> 16) & 1u);                  // round-to-nearest-even
  unsigned short h = (unsigned short)(u >> 16);
  return __builtin_bit_cast(__bf16, h);
}

static __device__ __forceinline__ v4bf pk4(f4 v) {
#if __has_builtin(__builtin_amdgcn_cvt_pk_bf16_f32)
  v2bf lo = __builtin_bit_cast(v2bf, __builtin_amdgcn_cvt_pk_bf16_f32(v.x, v.y));
  v2bf hi = __builtin_bit_cast(v2bf, __builtin_amdgcn_cvt_pk_bf16_f32(v.z, v.w));
  return __builtin_shufflevector(lo, hi, 0, 1, 2, 3);
#else
  return (v4bf){f2bf(v.x), f2bf(v.y), f2bf(v.z), f2bf(v.w)};
#endif
}

// ---------------- generic strided batched GEMM via WMMA bf16 ----------------
// C[b,h] = act( A[b,h](MxK) * B[b,h](KxN) + bias ), fp32 in/out, bf16 compute.
// Assumptions (hold for every call below): A K-stride == 1, K % 32 == 0,
// btrans=1 -> B K-stride == 1 (per-N stride = bcs); btrans=0 -> B N-stride == 1 (per-K stride = brs).
struct GemmP {
  const float* A; long long ars, absb, absh;
  const float* B; long long brs, bcs, bbsb, bbsh;
  float*       C; long long crs, ccs, cbsb, cbsh;
  const float* bias;
  int M, N, K, Hdim, act, btrans;                   // act: 0 none, 1 exact GELU
};

#define TBM 128
#define TBN 128
#define TBK 32
#define LPAD 8   // pad rows to 40 halves = 80B: 16B-aligned chunks, 20-bank stride (conflict-free)

__global__ __launch_bounds__(256) void txl_gemm_bf16(GemmP p) {
  __shared__ __attribute__((aligned(16))) __bf16 As[TBM][TBK + LPAD];
  __shared__ __attribute__((aligned(16))) __bf16 Bt[TBN][TBK + LPAD];  // transposed: [n][k]

  const int tid  = threadIdx.x;
  const int lane = tid & 31, wave = tid >> 5;
  const int lo   = lane & 15, hi = lane >> 4;
  const int wm   = wave >> 1, wn = wave & 1;        // waves tiled 4(M) x 2(N)

  const int bb = (int)blockIdx.z / p.Hdim;
  const int hh = (int)blockIdx.z % p.Hdim;
  gcf Ab = (gcf)p.A + (long long)bb * p.absb + (long long)hh * p.absh;
  gcf Bb = (gcf)p.B + (long long)bb * p.bbsb + (long long)hh * p.bbsh;
  gf  Cb = (gf)p.C  + (long long)bb * p.cbsb + (long long)hh * p.cbsh;

  const long long m0 = (long long)blockIdx.y * TBM;
  const long long n0 = (long long)blockIdx.x * TBN;

  // ---- precompute per-thread staging slots (4 float4 chunks per matrix) ----
  gcf aptr[4]; v4bf* asl[4]; bool aval[4];
  #pragma unroll
  for (int i = 0; i < 4; ++i) {
    int id = tid + 256 * i;                   // 1024 chunks = 128 rows x 8 chunks
    int r = id >> 3, c4 = (id & 7) * 4;
    long long gm = m0 + r;
    aval[i] = gm < p.M;
    aptr[i] = Ab + (aval[i] ? gm * p.ars : 0) + c4;
    asl[i]  = (v4bf*)&As[r][c4];
  }

  gcf bptr[4];
  v4bf* bsl[4];           // btrans=1 slots
  __bf16* bscat[4];       // btrans=0 slots (&Bt[n4][kk])
  bool bval[4]; int brem[4];
  #pragma unroll
  for (int i = 0; i < 4; ++i) {
    int id = tid + 256 * i;
    if (p.btrans) {                           // B[k][n], k contiguous: chunk = 4 k's of one n
      int r = id >> 3, c4 = (id & 7) * 4;
      long long gn = n0 + r;
      bval[i] = gn < p.N;
      bptr[i] = Bb + (bval[i] ? gn * p.bcs : 0) + c4;
      bsl[i]  = (v4bf*)&Bt[r][c4];
      brem[i] = 4;
    } else {                                  // B[k][n], n contiguous: chunk = 4 n's of one k
      int kk = id >> 5, n4 = (id & 31) * 4;
      long long gn = n0 + n4;
      long long rem = (long long)p.N - gn;    // how many of the 4 are valid
      brem[i] = rem >= 4 ? 4 : (rem < 0 ? 0 : (int)rem);
      bval[i] = brem[i] == 4;
      bptr[i] = Bb + (brem[i] ? gn : 0);      // k-term added in loop
      bscat[i] = &Bt[n4][kk];
      bsl[i] = nullptr;
    }
  }

  v8f acc[2][4];
  #pragma unroll
  for (int a = 0; a < 2; ++a)
    #pragma unroll
    for (int c = 0; c < 4; ++c) acc[a][c] = (v8f){0,0,0,0,0,0,0,0};

  const int nk = p.K / TBK;
  const long long bstep = p.btrans ? TBK : TBK * p.brs;

  for (int kt = 0; kt < nk; ++kt) {
    // ---- stage A tile: global_load_b128 -> pack bf16 -> ds_store_b64 ----
    #pragma unroll
    for (int i = 0; i < 4; ++i) {
      f4 v = *(gcf4)aptr[i];                       // always-valid address (row 0 if OOB)
      __builtin_prefetch(aptr[i] + TBK, 0, 1);     // speculative; dropped past end
      aptr[i] += TBK;
      if (!aval[i]) v = (f4){0.f, 0.f, 0.f, 0.f};
      *asl[i] = pk4(v);
    }
    // ---- stage B tile into transposed LDS ----
    if (p.btrans) {
      #pragma unroll
      for (int i = 0; i < 4; ++i) {
        f4 v = *(gcf4)bptr[i];
        __builtin_prefetch(bptr[i] + TBK, 0, 1);
        bptr[i] += bstep;
        if (!bval[i]) v = (f4){0.f, 0.f, 0.f, 0.f};
        *bsl[i] = pk4(v);
      }
    } else {
      #pragma unroll
      for (int i = 0; i < 4; ++i) {
        f4 v = (f4){0.f, 0.f, 0.f, 0.f};
        if (bval[i]) {
          v = *(gcf4)bptr[i];
          __builtin_prefetch(bptr[i] + TBK * p.brs, 0, 1);
        } else {
          gcf bp = bptr[i];
          if (brem[i] > 0) v.x = bp[0];
          if (brem[i] > 1) v.y = bp[1];
          if (brem[i] > 2) v.z = bp[2];
        }
        bptr[i] += bstep;
        v4bf h = pk4(v);
        __bf16* d = bscat[i];                 // scatter 4 N-rows at column kk
        d[0]               = h[0];
        d[TBK + LPAD]      = h[1];
        d[2*(TBK + LPAD)]  = h[2];
        d[3*(TBK + LPAD)]  = h[3];
      }
    }
    __syncthreads();

    // ---- WMMA: fragments are contiguous 16B runs -> ds_load_b128 ----
    #pragma unroll
    for (int mt = 0; mt < 2; ++mt) {
      const v8bf* ar = (const v8bf*)&As[wm * 32 + mt * 16 + lo][0];
      v8bf a0 = ar[hi];        // halves 8*hi .. 8*hi+7       (K 0..15 split by lane half)
      v8bf a1 = ar[2 + hi];    // halves 16+8*hi .. 16+8*hi+7 (K 16..31)
      v16bf a = __builtin_shufflevector(a0, a1, 0,1,2,3,4,5,6,7,8,9,10,11,12,13,14,15);
      #pragma unroll
      for (int nt = 0; nt < 4; ++nt) {
        const v8bf* br = (const v8bf*)&Bt[wn * 64 + nt * 16 + lo][0];
        v8bf b0 = br[hi];          // K 8*hi .. 8*hi+7
        v8bf b1 = br[2 + hi];      // K 16+8*hi .. 16+8*hi+7
        v16bf b = __builtin_shufflevector(b0, b1, 0,1,2,3,4,5,6,7,8,9,10,11,12,13,14,15);
        acc[mt][nt] = __builtin_amdgcn_wmma_f32_16x16x32_bf16(
            false, a, false, b, (short)0, acc[mt][nt], false, false);
      }
    }
    __syncthreads();
  }

  // ---- epilogue: C/D layout — VGPR j holds M = j + 8*hi, lane_lo = N ----
  gcf bias = (gcf)p.bias;
  #pragma unroll
  for (int mt = 0; mt < 2; ++mt)
    #pragma unroll
    for (int nt = 0; nt < 4; ++nt)
      #pragma unroll
      for (int j = 0; j < 8; ++j) {
        long long m = m0 + wm * 32 + mt * 16 + j + 8 * hi;
        long long n = n0 + wn * 64 + nt * 16 + lo;
        if (m < p.M && n < p.N) {
          float v = acc[mt][nt][j];
          if (p.bias) v += bias[n];
          if (p.act == 1) v = 0.5f * v * (1.0f + erff(v * 0.70710678118654752f));
          Cb[m * p.crs + n * p.ccs] = v;
        }
      }
}

// ---------------- embedding gather: core = emb[src] * sqrt(D) ----------------
__global__ __launch_bounds__(256) void txl_embed(float* core, const int* src, const float* emb) {
  long long idx = (long long)blockIdx.x * 256 + threadIdx.x;   // NTOK*DMODEL total
  long long row = idx >> 10;
  int d = (int)(idx & 1023);
  ((gf)core)[idx] = ((gcf)emb)[(long long)src[row] * DMODEL + d] * 32.0f;  // sqrt(1024)=32
}

// ---------------- sinusoidal relative position embedding (RLEN x D) ----------------
__global__ __launch_bounds__(256) void txl_posemb(float* r) {
  int idx = blockIdx.x * 256 + threadIdx.x;                    // RLEN * 512
  if (idx >= RLEN * (DMODEL / 2)) return;
  int p = idx / (DMODEL / 2), d2 = idx % (DMODEL / 2);
  float invf = expf(-(2.0f * (float)d2 / (float)DMODEL) * 9.210340371976184f); // ln(1e4)
  float pos  = (float)(Q_LEN - p);                             // pos_seq = klen..0
  float a = pos * invf;
  gf rg = (gf)r;
  rg[(long long)p * DMODEL + d2]              = sinf(a);
  rg[(long long)p * DMODEL + DMODEL / 2 + d2] = cosf(a);
}

// ---------------- q_bias: wq += rq[-1]  (in-place on qkv cols 0..D-1) ----------------
__global__ __launch_bounds__(256) void txl_qbias(float* qkv, const float* rproj) {
  long long idx = (long long)blockIdx.x * 256 + threadIdx.x;   // NTOK*DMODEL
  long long row = idx >> 10;
  int c = (int)(idx & 1023);
  ((gf)qkv)[row * (3 * DMODEL) + c] += ((gcf)rproj)[(long long)Q_LEN * 3 * DMODEL + c];
}

// ---------------- masked softmax with analytic rel-shift ----------------
// prob[t,i,j] = softmax_j( (AC[t,i,j] + BD[t,i,j-i+klen]) * SCALE ), j<=i
__global__ __launch_bounds__(512) void txl_softmax(float* ac, const float* bd) {
  __shared__ float red[512];
  const int i = blockIdx.x;
  const long long t = blockIdx.y;
  const int j = threadIdx.x;
  gf row = (gf)ac + (t * Q_LEN + i) * (long long)Q_LEN;
  gcf bdr = (gcf)bd + (t * Q_LEN + i) * (long long)RLEN;
  float s = -INFINITY;
  if (j <= i) s = (row[j] + bdr[j - i + Q_LEN]) * 0.125f;      // 1/sqrt(64)
  red[j] = s; __syncthreads();
  for (int o = 256; o > 0; o >>= 1) { if (j < o) red[j] = fmaxf(red[j], red[j + o]); __syncthreads(); }
  float mx = red[0]; __syncthreads();
  float e = (j <= i) ? expf(s - mx) : 0.f;
  red[j] = e; __syncthreads();
  for (int o = 256; o > 0; o >>= 1) { if (j < o) red[j] += red[j + o]; __syncthreads(); }
  row[j] = e / red[0];
}

// ---------------- fused residual + LayerNorm: out = LN(x + res)*g + b ----------------
__global__ __launch_bounds__(256) void txl_ln(float* out, const float* x, const float* res,
                                              const float* g, const float* b) {
  __shared__ float s1[256], s2[256];
  const long long row = blockIdx.x;
  gcf xr = (gcf)x   + row * DMODEL;
  gcf rr = (gcf)res + row * DMODEL;
  float v[4], s = 0.f, q = 0.f;
  #pragma unroll
  for (int k = 0; k < 4; ++k) {
    int d = threadIdx.x + k * 256;
    v[k] = xr[d] + rr[d]; s += v[k]; q += v[k] * v[k];
  }
  s1[threadIdx.x] = s; s2[threadIdx.x] = q; __syncthreads();
  for (int o = 128; o > 0; o >>= 1) {
    if (threadIdx.x < o) { s1[threadIdx.x] += s1[threadIdx.x + o]; s2[threadIdx.x] += s2[threadIdx.x + o]; }
    __syncthreads();
  }
  float mean = s1[0] * (1.0f / DMODEL);
  float var  = s2[0] * (1.0f / DMODEL) - mean * mean;
  float inv  = rsqrtf(var + 1e-5f);
  #pragma unroll
  for (int k = 0; k < 4; ++k) {
    int d = threadIdx.x + k * 256;
    ((gf)out)[row * DMODEL + d] = (v[k] - mean) * inv * ((gcf)g)[d] + ((gcf)b)[d];
  }
}

// ---------------- host side ----------------
static inline void launch_gemm(hipStream_t s,
    const float* A, long long ars, long long absb, long long absh,
    const float* B, long long brs, long long bcs, long long bbsb, long long bbsh,
    float* C, long long crs, long long ccs, long long cbsb, long long cbsh,
    const float* bias, int M, int N, int K, int batch, int Hdim, int act, int btrans) {
  GemmP p{A, ars, absb, absh, B, brs, bcs, bbsb, bbsh,
          C, crs, ccs, cbsb, cbsh, bias, M, N, K, Hdim, act, btrans};
  dim3 g((N + TBN - 1) / TBN, (M + TBM - 1) / TBM, batch);
  txl_gemm_bf16<<<g, 256, 0, s>>>(p);
}

extern "C" void kernel_launch(void* const* d_in, const int* in_sizes, int n_in,
                              void* d_out, int out_size, void* d_ws, size_t ws_size,
                              hipStream_t stream) {
  (void)in_sizes; (void)n_in; (void)out_size; (void)ws_size;
  const int*   src   = (const int*)  d_in[0];
  const float* emb   = (const float*)d_in[1];
  const float* dec_b = (const float*)d_in[2];
  const float* Wqkv  = (const float*)d_in[3];
  const float* Wo    = (const float*)d_in[4];
  const float* ln0g  = (const float*)d_in[5];
  const float* ln0b  = (const float*)d_in[6];
  const float* W1    = (const float*)d_in[7];
  const float* b1    = (const float*)d_in[8];
  const float* W2    = (const float*)d_in[9];
  const float* b2    = (const float*)d_in[10];
  const float* ln1g  = (const float*)d_in[11];
  const float* ln1b  = (const float*)d_in[12];
  const float* ln2g  = (const float*)d_in[13];
  const float* ln2b  = (const float*)d_in[14];
  float* logits = (float*)d_out;
  float* ws     = (float*)d_ws;

  // workspace layout (floats); FFN buffers alias attention score buffers
  const long long OFF_CORE  = 0;
  const long long OFF_R     = OFF_CORE  + (long long)NTOK * DMODEL;
  const long long OFF_QKV   = OFF_R     + (long long)RLEN * DMODEL;
  const long long OFF_RPROJ = OFF_QKV   + (long long)NTOK * 3 * DMODEL;
  const long long OFF_AC    = OFF_RPROJ + (long long)RLEN * 3 * DMODEL;
  const long long OFF_BD    = OFF_AC    + (long long)BHB * Q_LEN * Q_LEN;
  const long long OFF_VEC   = OFF_BD    + (long long)BHB * Q_LEN * RLEN;
  const long long OFF_ATTN  = OFF_VEC   + (long long)NTOK * DMODEL;
  const long long OFF_SRC2  = OFF_ATTN  + (long long)NTOK * DMODEL;
  const long long OFF_H     = OFF_SRC2  + (long long)NTOK * DMODEL;

  float* core  = ws + OFF_CORE;
  float* rbuf  = ws + OFF_R;
  float* qkv   = ws + OFF_QKV;
  float* rproj = ws + OFF_RPROJ;
  float* acbuf = ws + OFF_AC;     // also prob; also FFN mid (disjoint lifetime)
  float* bdbuf = ws + OFF_BD;     // also FFN out (disjoint lifetime)
  float* vecb  = ws + OFF_VEC;
  float* attnb = ws + OFF_ATTN;
  float* src2  = ws + OFF_SRC2;
  float* hbuf  = ws + OFF_H;
  float* ffmid = acbuf;
  float* ffout = bdbuf;

  txl_embed <<<(NTOK * DMODEL) / 256, 256, 0, stream>>>(core, src, emb);
  txl_posemb<<<(RLEN * (DMODEL / 2) + 255) / 256, 256, 0, stream>>>(rbuf);

  for (int l = 0; l < NLAYER; ++l) {
    const float* Wqkv_l = Wqkv + (long long)l * DMODEL * 3 * DMODEL;

    // qkv = core @ Wqkv[l]   (4096 x 3072)
    launch_gemm(stream, core, DMODEL, 0, 0,
                Wqkv_l, 3 * DMODEL, 1, 0, 0,
                qkv, 3 * DMODEL, 1, 0, 0,
                nullptr, NTOK, 3 * DMODEL, DMODEL, 1, 1, 0, /*btrans*/0);
    // rproj = r @ Wqkv[l]    (513 x 3072)
    launch_gemm(stream, rbuf, DMODEL, 0, 0,
                Wqkv_l, 3 * DMODEL, 1, 0, 0,
                rproj, 3 * DMODEL, 1, 0, 0,
                nullptr, RLEN, 3 * DMODEL, DMODEL, 1, 1, 0, 0);
    // wq += rq[-1]  -> q_bias in-place
    txl_qbias<<<(NTOK * DMODEL) / 256, 256, 0, stream>>>(qkv, rproj);

    // AC[t,i,j] = q_bias . wk   (batched over b,h; wk K-contiguous -> btrans=1)
    launch_gemm(stream,
        qkv, (long long)BSZ * 3 * DMODEL, 3 * DMODEL, DHEAD,
        qkv + DMODEL, 1, (long long)BSZ * 3 * DMODEL, 3 * DMODEL, DHEAD,
        acbuf, Q_LEN, 1, (long long)NHEAD * Q_LEN * Q_LEN, (long long)Q_LEN * Q_LEN,
        nullptr, Q_LEN, Q_LEN, DHEAD, BHB, NHEAD, 0, 1);
    // BD[t,i,p] = q_bias . rk   (rk shared across b; btrans=1)
    launch_gemm(stream,
        qkv, (long long)BSZ * 3 * DMODEL, 3 * DMODEL, DHEAD,
        rproj + DMODEL, 1, 3 * DMODEL, 0, DHEAD,
        bdbuf, RLEN, 1, (long long)NHEAD * Q_LEN * RLEN, (long long)Q_LEN * RLEN,
        nullptr, Q_LEN, RLEN, DHEAD, BHB, NHEAD, 0, 1);

    // masked softmax with analytic rel-shift; prob overwrites acbuf
    txl_softmax<<<dim3(Q_LEN, BHB), 512, 0, stream>>>(acbuf, bdbuf);

    // vec = prob @ wv  (wv N-contiguous -> btrans=0)
    launch_gemm(stream,
        acbuf, Q_LEN, (long long)NHEAD * Q_LEN * Q_LEN, (long long)Q_LEN * Q_LEN,
        qkv + 2 * DMODEL, (long long)BSZ * 3 * DMODEL, 1, 3 * DMODEL, DHEAD,
        vecb, (long long)BSZ * DMODEL, 1, DMODEL, DHEAD,
        nullptr, Q_LEN, DHEAD, Q_LEN, BHB, NHEAD, 0, 0);

    // attn_out = vec @ Wo[l]
    launch_gemm(stream, vecb, DMODEL, 0, 0,
                Wo + (long long)l * DMODEL * DMODEL, DMODEL, 1, 0, 0,
                attnb, DMODEL, 1, 0, 0,
                nullptr, NTOK, DMODEL, DMODEL, 1, 1, 0, 0);

    // src2 = LN(core + attn_out); h = LN(core + src2)
    txl_ln<<<NTOK, 256, 0, stream>>>(src2, core, attnb, ln0g + l * DMODEL, ln0b + l * DMODEL);
    txl_ln<<<NTOK, 256, 0, stream>>>(hbuf, core, src2, ln1g + l * DMODEL, ln1b + l * DMODEL);

    // FFN: gelu(h @ W1 + b1) @ W2 + b2
    launch_gemm(stream, hbuf, DMODEL, 0, 0,
                W1 + (long long)l * DMODEL * FFDIM, FFDIM, 1, 0, 0,
                ffmid, FFDIM, 1, 0, 0,
                b1 + (long long)l * FFDIM, NTOK, FFDIM, DMODEL, 1, 1, 1, 0);
    launch_gemm(stream, ffmid, FFDIM, 0, 0,
                W2 + (long long)l * FFDIM * DMODEL, DMODEL, 1, 0, 0,
                ffout, DMODEL, 1, 0, 0,
                b2 + (long long)l * DMODEL, NTOK, DMODEL, FFDIM, 1, 1, 0, 0);

    // core = LN(h + ffn)
    txl_ln<<<NTOK, 256, 0, stream>>>(core, hbuf, ffout, ln2g + l * DMODEL, ln2b + l * DMODEL);
  }

  // logits = core @ emb^T + dec_b   (emb rows are the K dim -> btrans=1)
  launch_gemm(stream, core, DMODEL, 0, 0,
              emb, 1, DMODEL, 0, 0,
              logits, VOCAB, 1, 0, 0,
              dec_b, NTOK, VOCAB, DMODEL, 1, 1, 0, 1);
}
